// DampedAttention_515396075660
// MI455X (gfx1250) — matrix operand order repaired and checked by reading, not compiled, not run on hardware
//
#include <hip/hip_runtime.h>
#include <hip/hip_bf16.h>
#include <cstdint>

// ---------------------------------------------------------------------------
// DampedAttention forward for MI455X (gfx1250, wave32, WMMA).
// Pipeline: f32->bf16 convert | WMMA GEMM (Q,K,V proj, async-LDS staged B) |
//           fused softmax-attn | banded-diffusion + bf16 convert | WMMA GEMM.
// Workspace use: 64 MB.
// ---------------------------------------------------------------------------

#define BSZ      2
#define SEQ      2048
#define DIMM     1024
#define NHEADS   16
#define HD       64
#define MROWS    (BSZ * SEQ)     // 4096

typedef __attribute__((ext_vector_type(16))) __bf16         bf16x16;
typedef __attribute__((ext_vector_type(8)))  float          f32x8;
typedef __attribute__((ext_vector_type(8)))  unsigned short u16x8;
typedef __attribute__((ext_vector_type(16))) unsigned short u16x16;

union HalfPack { u16x16 v; u16x8 h[2]; };

static __device__ __forceinline__ bf16x16 as_bf(u16x16 v) {
  return __builtin_bit_cast(bf16x16, v);
}

static __device__ __forceinline__ unsigned short f2bf(float f) {
  unsigned u = __float_as_uint(f);
  u = (u + 0x7FFFu + ((u >> 16) & 1u)) >> 16;   // round-to-nearest-even
  return (unsigned short)u;
}
static __device__ __forceinline__ float bf2f(unsigned short h) {
  return __uint_as_float(((unsigned)h) << 16);
}

// Async copy of one 32-byte chunk (global -> LDS), per lane.
// INST_OFFSET is added to both global and LDS addresses (ISA 15.18.3).
static __device__ __forceinline__ void async_copy32(unsigned lds_off,
                                                    const unsigned short* g) {
  asm volatile(
      "global_load_async_to_lds_b128 %0, %1, off\n\t"
      "global_load_async_to_lds_b128 %0, %1, off offset:16"
      :: "v"(lds_off), "v"(g) : "memory");
}

// ---------------------------------------------------------------------------
// f32 -> bf16 convert
// ---------------------------------------------------------------------------
__global__ void cvt_f32_bf16(const float* __restrict__ src,
                             unsigned short* __restrict__ dst, int n) {
  int i = blockIdx.x * blockDim.x + threadIdx.x;
  if (i < n) dst[i] = f2bf(src[i]);
}

// ---------------------------------------------------------------------------
// Y = X (MxK, bf16) @ W^T (W is NxK, bf16) + bias (f32).
// Block = 128 threads (4 waves). Block tile 128x64; wave tile 32x64.
// B tile (64x32 bf16 = 4KB) staged in LDS via async-to-LDS, double-buffered.
// mode 0: f32 row-major [M,N] to dst
// mode 1: bf16 to [B*H, S, HD]  (value scaled by `scale`)  -- Q / K
// mode 2: bf16 to [B*H, HD, S]  (per-head transposed)       -- V
// ---------------------------------------------------------------------------
__global__ __launch_bounds__(128)
void gemm_bf16(const unsigned short* __restrict__ X,
               const unsigned short* __restrict__ W,
               const float* __restrict__ bias,
               int M, int N, int K, int mode, float scale,
               void* __restrict__ dst) {
  __shared__ unsigned short lds_b[2][64 * 32];   // double-buffered B tile

  const int t    = threadIdx.x;
  const int wave = t >> 5;
  const int L    = t & 31;
  const int lm   = L & 15;          // A-row / B-col within 16-tile
  const int half = L >> 4;          // K-half selector
  const int m0   = blockIdx.x * 128 + wave * 32;
  const int n0   = blockIdx.y * 64;

  // Cooperative B staging: thread t owns row (t>>1), 32B chunk (t&1).
  const unsigned short* wsrc = W + (size_t)(n0 + (t >> 1)) * K + (t & 1) * 16;
  const unsigned lbase = (unsigned)(uintptr_t)&lds_b[0][(t >> 1) * 32 + (t & 1) * 16];
  const unsigned lstep = (unsigned)(64 * 32 * 2);   // bytes per buffer

  f32x8 acc_lo[4] = {};
  f32x8 acc_hi[4] = {};
  const unsigned short* xlo = X + (size_t)(m0 + lm) * K;
  const unsigned short* xhi = X + (size_t)(m0 + 16 + lm) * K;

  // Prologue: stage K-block 0 into buffer 0.
  async_copy32(lbase, wsrc);

  int cur = 0;
  int k0 = 0;
  for (; k0 + 32 < K; k0 += 32) {
    // Stage next K-block into the other buffer.
    async_copy32(lbase + (unsigned)(cur ^ 1) * lstep, wsrc + k0 + 32);
    // Previous batch (2 async ops) is the oldest; async loads retire in
    // order, so <=2 outstanding means the current buffer has landed.
    asm volatile("s_wait_asynccnt 0x2" ::: "memory");
    __syncthreads();

    HalfPack u;
    u.h[0] = *(const u16x8*)(xlo + k0 + half * 16);
    u.h[1] = *(const u16x8*)(xlo + k0 + half * 16 + 8);
    bf16x16 a0 = as_bf(u.v);
    u.h[0] = *(const u16x8*)(xhi + k0 + half * 16);
    u.h[1] = *(const u16x8*)(xhi + k0 + half * 16 + 8);
    bf16x16 a1 = as_bf(u.v);
#pragma unroll
    for (int nt = 0; nt < 4; ++nt) {
      bf16x16 bv =
          as_bf(*(const u16x16*)&lds_b[cur][(nt * 16 + lm) * 32 + half * 16]);
      acc_lo[nt] = __builtin_amdgcn_wmma_f32_16x16x32_bf16(
          false, a0, false, bv, (short)0, acc_lo[nt], false, false);
      acc_hi[nt] = __builtin_amdgcn_wmma_f32_16x16x32_bf16(
          false, a1, false, bv, (short)0, acc_hi[nt], false, false);
    }
    __syncthreads();   // reads of buffer `cur` done before it is re-staged
    cur ^= 1;
  }
  // Tail K-block.
  asm volatile("s_wait_asynccnt 0x0" ::: "memory");
  __syncthreads();
  {
    HalfPack u;
    u.h[0] = *(const u16x8*)(xlo + k0 + half * 16);
    u.h[1] = *(const u16x8*)(xlo + k0 + half * 16 + 8);
    bf16x16 a0 = as_bf(u.v);
    u.h[0] = *(const u16x8*)(xhi + k0 + half * 16);
    u.h[1] = *(const u16x8*)(xhi + k0 + half * 16 + 8);
    bf16x16 a1 = as_bf(u.v);
#pragma unroll
    for (int nt = 0; nt < 4; ++nt) {
      bf16x16 bv =
          as_bf(*(const u16x16*)&lds_b[cur][(nt * 16 + lm) * 32 + half * 16]);
      acc_lo[nt] = __builtin_amdgcn_wmma_f32_16x16x32_bf16(
          false, a0, false, bv, (short)0, acc_lo[nt], false, false);
      acc_hi[nt] = __builtin_amdgcn_wmma_f32_16x16x32_bf16(
          false, a1, false, bv, (short)0, acc_hi[nt], false, false);
    }
  }

  // Epilogue.
#pragma unroll
  for (int hh = 0; hh < 2; ++hh) {
    const f32x8* acc = hh ? acc_hi : acc_lo;
#pragma unroll
    for (int nt = 0; nt < 4; ++nt) {
      const int n = n0 + nt * 16 + lm;
      const float bv = bias[n];
#pragma unroll
      for (int r = 0; r < 8; ++r) {
        const int m = m0 + hh * 16 + r + 8 * half;
        const float y = (acc[nt][r] + bv) * scale;
        if (mode == 0) {
          ((float*)dst)[(size_t)m * N + n] = y;
        } else {
          const int b = m / SEQ, s = m % SEQ;
          const int h = n >> 6, d = n & 63;
          if (mode == 1) {
            ((unsigned short*)dst)[((size_t)(b * NHEADS + h) * SEQ + s) * HD + d] =
                f2bf(y);
          } else {
            ((unsigned short*)dst)[((size_t)(b * NHEADS + h) * HD + d) * SEQ + s] =
                f2bf(y);
          }
        }
      }
    }
  }
}

// ---------------------------------------------------------------------------
// Attention: one wave per (b,h, 16-query tile).
// Q,K: bf16 [BH, S, HD];  Vt: bf16 [BH, HD, S];  O: f32 [B, S, DIM].
// Q already scaled by 1/(tau*sqrt(hd)).  Writes 0.6 * softmax(QK^T) @ V.
// ---------------------------------------------------------------------------
static __device__ __forceinline__
f32x8 qk_tile(const unsigned short* __restrict__ Kb, int key0,
              int lm, int half, bf16x16 aq0, bf16x16 aq1) {
  const unsigned short* kr = Kb + (size_t)(key0 + lm) * HD + half * 16;
  bf16x16 b0 = as_bf(*(const u16x16*)kr);           // d = half*16 .. +16
  bf16x16 b1 = as_bf(*(const u16x16*)(kr + 32));    // d = 32+half*16 .. +16
  f32x8 c = {};
  c = __builtin_amdgcn_wmma_f32_16x16x32_bf16(false, aq0, false, b0,
                                              (short)0, c, false, false);
  c = __builtin_amdgcn_wmma_f32_16x16x32_bf16(false, aq1, false, b1,
                                              (short)0, c, false, false);
  return c;
}

__global__ __launch_bounds__(32)
void attn_kernel(const unsigned short* __restrict__ Q,
                 const unsigned short* __restrict__ Km,
                 const unsigned short* __restrict__ Vt,
                 float* __restrict__ O) {
  __shared__ unsigned short lp[16 * 32];   // 16x32 P tile (bf16)

  const int bh = blockIdx.x >> 7;          // SEQ/16 = 128 tiles per head
  const int q0 = (blockIdx.x & 127) << 4;
  const int L = threadIdx.x, lm = L & 15, half = L >> 4;

  const unsigned short* Qb = Q  + (size_t)bh * SEQ * HD;
  const unsigned short* Kb = Km + (size_t)bh * SEQ * HD;
  const unsigned short* Vb = Vt + (size_t)bh * HD * SEQ;

  // Q A-operand (16 x 64 -> two K=32 operands), loaded once.
  bf16x16 aq0, aq1;
  {
    const unsigned short* qrow = Qb + (size_t)(q0 + lm) * HD;
    HalfPack u;
    u.h[0] = *(const u16x8*)(qrow + half * 16);
    u.h[1] = *(const u16x8*)(qrow + half * 16 + 8);
    aq0 = as_bf(u.v);
    u.h[0] = *(const u16x8*)(qrow + 32 + half * 16);
    u.h[1] = *(const u16x8*)(qrow + 40 + half * 16);
    aq1 = as_bf(u.v);
  }

  // Pass 1: global row max (rows r + 8*half for this lane).
  float mrow[8];
#pragma unroll
  for (int r = 0; r < 8; ++r) mrow[r] = -1e30f;
  for (int c = 0; c < SEQ / 32; ++c) {
    const int k0 = c * 32;
    f32x8 s0 = qk_tile(Kb, k0,      lm, half, aq0, aq1);
    f32x8 s1 = qk_tile(Kb, k0 + 16, lm, half, aq0, aq1);
#pragma unroll
    for (int r = 0; r < 8; ++r)
      mrow[r] = fmaxf(mrow[r], fmaxf(s0[r], s1[r]));
  }
#pragma unroll
  for (int r = 0; r < 8; ++r)
#pragma unroll
    for (int mask = 1; mask <= 8; mask <<= 1)
      mrow[r] = fmaxf(mrow[r], __shfl_xor(mrow[r], mask));

  // Pass 2: P = exp(s - max); accumulate O = P @ V and row sums.
  float lsum[8] = {0.f, 0.f, 0.f, 0.f, 0.f, 0.f, 0.f, 0.f};
  f32x8 o[4] = {};
  for (int c = 0; c < SEQ / 32; ++c) {
    const int k0 = c * 32;
    f32x8 s0 = qk_tile(Kb, k0,      lm, half, aq0, aq1);
    f32x8 s1 = qk_tile(Kb, k0 + 16, lm, half, aq0, aq1);
#pragma unroll
    for (int r = 0; r < 8; ++r) {
      const float p0 = __expf(s0[r] - mrow[r]);
      const float p1 = __expf(s1[r] - mrow[r]);
      lsum[r] += p0 + p1;
      const int row = r + 8 * half;
      lp[row * 32 + lm]      = f2bf(p0);
      lp[row * 32 + 16 + lm] = f2bf(p1);
    }
    __syncthreads();
    // Re-read P in A-operand layout.
    HalfPack u;
    u.h[0] = *(const u16x8*)&lp[lm * 32 + half * 16];
    u.h[1] = *(const u16x8*)&lp[lm * 32 + half * 16 + 8];
    bf16x16 ap = as_bf(u.v);
#pragma unroll
    for (int nt = 0; nt < 4; ++nt) {
      const unsigned short* vrow =
          Vb + (size_t)(nt * 16 + lm) * SEQ + k0 + half * 16;
      bf16x16 bv = as_bf(*(const u16x16*)vrow);
      o[nt] = __builtin_amdgcn_wmma_f32_16x16x32_bf16(
          false, ap, false, bv, (short)0, o[nt], false, false);
    }
    __syncthreads();
  }

#pragma unroll
  for (int r = 0; r < 8; ++r)
#pragma unroll
    for (int mask = 1; mask <= 8; mask <<= 1)
      lsum[r] += __shfl_xor(lsum[r], mask);

  const int b = bh >> 4, h = bh & 15;
#pragma unroll
  for (int nt = 0; nt < 4; ++nt) {
#pragma unroll
    for (int r = 0; r < 8; ++r) {
      const int s = q0 + r + 8 * half;
      const int d = h * HD + nt * 16 + lm;
      O[(size_t)(b * SEQ + s) * DIMM + d] = o[nt][r] * (0.6f / lsum[r]);
    }
  }
}

// ---------------------------------------------------------------------------
// Add 0.4 * (L @ V) (banded Gaussian, window 3, strength 0.4, row-normalized)
// and convert concatenated heads to bf16 for the Wo GEMM.
// ---------------------------------------------------------------------------
__global__ void diffuse_cvt(const float* __restrict__ attn,
                            const unsigned short* __restrict__ Vt,
                            unsigned short* __restrict__ out) {
  const int idx = blockIdx.x * blockDim.x + threadIdx.x;
  const int m = idx >> 10, n = idx & 1023;
  const int b = m >> 11, s = m & (SEQ - 1);
  const int h = n >> 6, d = n & 63;
  const unsigned short* vrow =
      Vt + ((size_t)(b * NHEADS + h) * HD + d) * SEQ;
  float num = 0.f, den = 0.f;
#pragma unroll
  for (int j = -3; j <= 3; ++j) {
    const int ss = s + j;
    const float w = __expf(-(float)(j * j) * 3.125f);  // 1/(2*0.4^2)=3.125
    if (ss >= 0 && ss < SEQ) {
      den += w;
      num += w * bf2f(vrow[ss]);
    }
  }
  const float local = num / (den + 1e-10f);
  out[idx] = f2bf(attn[idx] + 0.4f * local);
}

// ---------------------------------------------------------------------------
extern "C" void kernel_launch(void* const* d_in, const int* in_sizes, int n_in,
                              void* d_out, int out_size, void* d_ws,
                              size_t ws_size, hipStream_t stream) {
  (void)in_sizes; (void)n_in; (void)out_size; (void)ws_size;
  const float* x  = (const float*)d_in[0];
  const float* Wq = (const float*)d_in[1];
  const float* bq = (const float*)d_in[2];
  const float* Wk = (const float*)d_in[3];
  const float* bk = (const float*)d_in[4];
  const float* Wv = (const float*)d_in[5];
  const float* bv = (const float*)d_in[6];
  const float* Wo = (const float*)d_in[7];
  const float* bo = (const float*)d_in[8];

  // Workspace carve (all sizes 256B-aligned). Total = 64 MB.
  char* w = (char*)d_ws;
  unsigned short* xb   = (unsigned short*)w; w += (size_t)MROWS * DIMM * 2;  // 8MB
  unsigned short* wqb  = (unsigned short*)w; w += (size_t)DIMM * DIMM * 2;   // 2MB
  unsigned short* wkb  = (unsigned short*)w; w += (size_t)DIMM * DIMM * 2;
  unsigned short* wvb  = (unsigned short*)w; w += (size_t)DIMM * DIMM * 2;
  unsigned short* wob  = (unsigned short*)w; w += (size_t)DIMM * DIMM * 2;
  unsigned short* qs   = (unsigned short*)w; w += (size_t)MROWS * DIMM * 2;  // 8MB
  unsigned short* ks   = (unsigned short*)w; w += (size_t)MROWS * DIMM * 2;  // 8MB
  unsigned short* vts  = (unsigned short*)w; w += (size_t)MROWS * DIMM * 2;  // 8MB
  float*          attf = (float*)w;          w += (size_t)MROWS * DIMM * 4;  // 16MB
  unsigned short* attb = (unsigned short*)w; w += (size_t)MROWS * DIMM * 2;  // 8MB

  const int NX = MROWS * DIMM;     // 4194304
  const int NW = DIMM * DIMM;      // 1048576

  cvt_f32_bf16<<<NX / 256, 256, 0, stream>>>(x,  xb,  NX);
  cvt_f32_bf16<<<NW / 256, 256, 0, stream>>>(Wq, wqb, NW);
  cvt_f32_bf16<<<NW / 256, 256, 0, stream>>>(Wk, wkb, NW);
  cvt_f32_bf16<<<NW / 256, 256, 0, stream>>>(Wv, wvb, NW);
  cvt_f32_bf16<<<NW / 256, 256, 0, stream>>>(Wo, wob, NW);

  dim3 ggrid(MROWS / 128, DIMM / 64);
  // Q: fold softmax scale 1/(tau*sqrt(64)) = 0.125 into the projection.
  gemm_bf16<<<ggrid, 128, 0, stream>>>(xb, wqb, bq, MROWS, DIMM, DIMM,
                                       /*mode=*/1, 0.125f, qs);
  gemm_bf16<<<ggrid, 128, 0, stream>>>(xb, wkb, bk, MROWS, DIMM, DIMM,
                                       /*mode=*/1, 1.0f, ks);
  gemm_bf16<<<ggrid, 128, 0, stream>>>(xb, wvb, bv, MROWS, DIMM, DIMM,
                                       /*mode=*/2, 1.0f, vts);

  attn_kernel<<<BSZ * NHEADS * (SEQ / 16), 32, 0, stream>>>(qs, ks, vts, attf);

  diffuse_cvt<<<NX / 256, 256, 0, stream>>>(attf, vts, attb);

  gemm_bf16<<<ggrid, 128, 0, stream>>>(attb, wob, bo, MROWS, DIMM, DIMM,
                                       /*mode=*/0, 1.0f, (float*)d_out);
}
